// ModelSequence_76982993813740
// MI455X (gfx1250) — matrix operand order repaired and checked by reading, not compiled, not run on hardware
//
#include <hip/hip_runtime.h>
#include <math.h>
#include <stdint.h>

#define HDIM   512
#define NNODES 8192
#define NSTEPS 16

typedef __attribute__((ext_vector_type(16))) _Float16 v16h;
typedef __attribute__((ext_vector_type(8)))  _Float16 v8h;
typedef __attribute__((ext_vector_type(4)))  _Float16 v4h;
typedef __attribute__((ext_vector_type(8)))  float    v8f;
typedef __attribute__((ext_vector_type(4)))  int      v4i;

#define BLK_R 128          // batch rows per block
#define BLK_C 128          // feature cols per block
#define KSLAB 64           // K per staged slab (2 WMMA k-steps)
#define LDSB  72           // padded halfs per LDS row (144B): aligned + conflict-free
#define ABUF  (BLK_R * LDSB)

#if defined(__has_builtin)
#if __has_builtin(__builtin_amdgcn_global_load_async_to_lds_b128)
#define HAVE_ASYNC 1
#endif
#endif

typedef __attribute__((address_space(1))) v4i gv4i;   // global int4
typedef __attribute__((address_space(3))) v4i sv4i;   // LDS int4

__device__ __forceinline__ void wait_async() {
#if defined(HAVE_ASYNC)
#if __has_builtin(__builtin_amdgcn_s_wait_asynccnt)
  __builtin_amdgcn_s_wait_asynccnt(0);
#else
  asm volatile("s_wait_asynccnt 0x0" ::: "memory");
#endif
#endif
}

struct __align__(16) Smem {
  _Float16 A[2 * ABUF];   // double-buffered A tile [row][k0..63], padded
  _Float16 B[2 * ABUF];   // double-buffered Bt tile [col][k0..63], padded
};

__device__ __forceinline__ float sigmoidf(float x) {
  return 1.0f / (1.0f + __expf(-x));
}

// One K=64 slab: 2 k-steps x 8 WMMAs, fragments per CDNA5 16-bit layouts.
__device__ __forceinline__ void compute_slab(
    const _Float16* __restrict__ Ac, const _Float16* __restrict__ Bc,
    int wave_r, int wave_c, int lh, int hi, v8f acc[2][4])
{
#pragma unroll
  for (int ks = 0; ks < 2; ++ks) {
    const int ko = ks * 32;
    v16h afrag[2];
#pragma unroll
    for (int rt = 0; rt < 2; ++rt) {
      const _Float16* ap = Ac + (wave_r * 32 + rt * 16 + lh) * LDSB + ko + hi * 8;
      const v8h lo = *(const v8h*)ap;          // K 0-7   (or 8-15)
      const v8h hh = *(const v8h*)(ap + 16);   // K 16-23 (or 24-31)
      afrag[rt] = __builtin_shufflevector(lo, hh,
          0,1,2,3,4,5,6,7,8,9,10,11,12,13,14,15);
    }
#pragma unroll
    for (int ct = 0; ct < 4; ++ct) {
      const _Float16* bp = Bc + (wave_c * 64 + ct * 16 + lh) * LDSB + ko + hi * 8;
      const v8h lo = *(const v8h*)bp;
      const v8h hh = *(const v8h*)(bp + 16);
      const v16h bfrag = __builtin_shufflevector(lo, hh,
          0,1,2,3,4,5,6,7,8,9,10,11,12,13,14,15);
#pragma unroll
      for (int rt = 0; rt < 2; ++rt) {
        acc[rt][ct] = __builtin_amdgcn_wmma_f32_16x16x32_f16(
            false, afrag[rt], false, bfrag, (short)0, acc[rt][ct], false, false);
      }
    }
  }
}

// D(row0..+128, bcol0..+128) += concat_s(seg[s]) @ Bt^T, all operands f16.
// seg[s]: [NNODES][HDIM] f16. Bt: [M][ldb] f16 (k contiguous per output col).
template<int NSEG>
__device__ __forceinline__ void gemm_tile(
    const _Float16* __restrict__ seg0, const _Float16* __restrict__ seg1,
    const _Float16* __restrict__ seg2, const _Float16* __restrict__ seg3,
    const _Float16* __restrict__ Bt, int ldb, int bcol0, int row0,
    Smem& sm, v8f acc[2][4])
{
  const int tid    = threadIdx.x;
  const int lane   = tid & 31;
  const int wid    = tid >> 5;
  const int wave_r = wid & 3;
  const int wave_c = wid >> 2;
  const int lh     = lane & 15;
  const int hi     = lane >> 4;

  // staging: each thread moves four 16B chunks for A and four for B per slab
  const int crow = tid >> 3;          // 0..31
  const int k8   = (tid & 7) * 8;     // 0..56

  auto segp = [&](int s) -> const _Float16* {
    return (s == 0) ? seg0 : (s == 1) ? seg1 : (s == 2) ? seg2 : seg3;
  };

#if defined(HAVE_ASYNC)
#define GASYNC(srcp_, dstp_)                                                      \
  __builtin_amdgcn_global_load_async_to_lds_b128(                                 \
      (gv4i*)(uintptr_t)(srcp_),                                                  \
      (sv4i*)(uint32_t)(uintptr_t)(dstp_), 0, 0)

  // DMA one K=64 slab (A from segment `aseg`, slab kbInSeg; B at global slab kt)
  auto stage = [&](const _Float16* aseg, int kbInSeg, int ktGlob, int bsel) {
    const _Float16* a = aseg + (size_t)(row0 + crow) * HDIM + kbInSeg * KSLAB + k8;
    const _Float16* b = Bt + (size_t)(bcol0 + crow) * ldb + ktGlob * KSLAB + k8;
    _Float16* dA = sm.A + bsel * ABUF;
    _Float16* dB = sm.B + bsel * ABUF;
#pragma unroll
    for (int i = 0; i < 4; ++i) {
      const int d = (crow + 32 * i) * LDSB + k8;
      GASYNC(a + (size_t)(32 * i) * HDIM, dA + d);
      GASYNC(b + (size_t)(32 * i) * ldb,  dB + d);
    }
  };

  stage(seg0, 0, 0, 0);
  wait_async();
  __syncthreads();
#pragma unroll
  for (int s = 0; s < NSEG; ++s) {
    const _Float16* cs = segp(s);
#pragma unroll 1
    for (int kb = 0; kb < 7; ++kb) {
      const int cb = kb & 1;
      stage(cs, kb + 1, s * 8 + kb + 1, cb ^ 1);            // DMA next slab
      compute_slab(sm.A + cb * ABUF, sm.B + cb * ABUF,       // math on current
                   wave_r, wave_c, lh, hi, acc);
      wait_async();
      __syncthreads();
    }
    // last slab of this segment (buffer 1); pre-stage next segment's slab 0
    if (s + 1 < NSEG) stage(segp(s + 1), 0, (s + 1) * 8, 0);
    compute_slab(sm.A + ABUF, sm.B + ABUF, wave_r, wave_c, lh, hi, acc);
    wait_async();
    __syncthreads();
  }
#undef GASYNC
#else
  v8h ra[4], rb[4];
  auto ldnext = [&](const _Float16* aseg, int kbInSeg, int ktGlob) {
    const _Float16* a = aseg + (size_t)(row0 + crow) * HDIM + kbInSeg * KSLAB + k8;
    const _Float16* b = Bt + (size_t)(bcol0 + crow) * ldb + ktGlob * KSLAB + k8;
#pragma unroll
    for (int i = 0; i < 4; ++i) {
      ra[i] = *(const v8h*)(a + (size_t)(32 * i) * HDIM);
      rb[i] = *(const v8h*)(b + (size_t)(32 * i) * ldb);
    }
  };
  auto stnext = [&](int bsel) {
    _Float16* dA = sm.A + bsel * ABUF;
    _Float16* dB = sm.B + bsel * ABUF;
#pragma unroll
    for (int i = 0; i < 4; ++i) {
      const int d = (crow + 32 * i) * LDSB + k8;
      *(v8h*)(dA + d) = ra[i];
      *(v8h*)(dB + d) = rb[i];
    }
  };

  ldnext(seg0, 0, 0);
  stnext(0);
  __syncthreads();
#pragma unroll
  for (int s = 0; s < NSEG; ++s) {
    const _Float16* cs = segp(s);
#pragma unroll 1
    for (int kb = 0; kb < 7; ++kb) {
      const int cb = kb & 1;
      ldnext(cs, kb + 1, s * 8 + kb + 1);
      compute_slab(sm.A + cb * ABUF, sm.B + cb * ABUF,
                   wave_r, wave_c, lh, hi, acc);
      stnext(cb ^ 1);
      __syncthreads();
    }
    const bool hn = (s + 1 < NSEG);
    if (hn) ldnext(segp(s + 1), 0, (s + 1) * 8);
    compute_slab(sm.A + ABUF, sm.B + ABUF, wave_r, wave_c, lh, hi, acc);
    if (hn) stnext(0);
    __syncthreads();
  }
#endif
}

// ---------------- GEMM kernels with fused epilogues ----------------

// Step 1: xi = sigmoid([hx,el,x0,xl]@W_xi + b); x0h=(1-xi)*x0; xth=xi*xl (f16 out)
__global__ void __launch_bounds__(256)
k_step1(const _Float16* __restrict__ hx16, const _Float16* __restrict__ el16,
        const _Float16* __restrict__ x016, const _Float16* __restrict__ xl16,
        const _Float16* __restrict__ Wt,  const float* __restrict__ bxi,
        const float* __restrict__ x0f, const float* __restrict__ xlf,
        _Float16* __restrict__ x0h16, _Float16* __restrict__ xth16)
{
  __shared__ Smem sm;
  const int row0 = blockIdx.x * BLK_R;
  const int col0 = blockIdx.y * BLK_C;
  v8f acc[2][4] = {};
  gemm_tile<4>(hx16, el16, x016, xl16, Wt, 4 * HDIM, col0, row0, sm, acc);

  const int lane = threadIdx.x & 31, wid = threadIdx.x >> 5;
  const int wave_r = wid & 3, wave_c = wid >> 2;
  const int lh = lane & 15, hi = lane >> 4;
#pragma unroll
  for (int ct = 0; ct < 4; ++ct) {
    const int m = col0 + wave_c * 64 + ct * 16 + lh;
    const float bias = bxi[m];
#pragma unroll
    for (int rt = 0; rt < 2; ++rt)
#pragma unroll
      for (int j = 0; j < 8; ++j) {
        const int n = row0 + wave_r * 32 + rt * 16 + hi * 8 + j;
        const size_t idx = (size_t)n * HDIM + m;
        const float xi = sigmoidf(acc[rt][ct][j] + bias);
        x0h16[idx] = (_Float16)((1.0f - xi) * x0f[idx]);
        xth16[idx] = (_Float16)(xi * xlf[idx]);
      }
  }
}

// Step 2: fused u/r with concatenated pre-transposed Wt_ur (M=1024, K=1536).
// cols [0,512): u = sigmoid (f32 out); [512,1024): rhx = r*hx (f16 out)
__global__ void __launch_bounds__(256)
k_step2(const _Float16* __restrict__ hx16, const _Float16* __restrict__ x0h16,
        const _Float16* __restrict__ xth16,
        const _Float16* __restrict__ Wt_ur,
        const float* __restrict__ bu, const float* __restrict__ br,
        const float* __restrict__ hxf,
        float* __restrict__ u, _Float16* __restrict__ rhx16)
{
  __shared__ Smem sm;
  const int row0 = blockIdx.x * BLK_R;
  const int col0 = blockIdx.y * BLK_C;      // 0..1023
  const bool isr = (col0 >= HDIM);
  v8f acc[2][4] = {};
  gemm_tile<3>(hx16, x0h16, xth16, nullptr, Wt_ur, 3 * HDIM, col0, row0, sm, acc);

  const int lane = threadIdx.x & 31, wid = threadIdx.x >> 5;
  const int wave_r = wid & 3, wave_c = wid >> 2;
  const int lh = lane & 15, hi = lane >> 4;
#pragma unroll
  for (int ct = 0; ct < 4; ++ct) {
    const int mg  = col0 + wave_c * 64 + ct * 16 + lh;
    const int ml  = isr ? (mg - HDIM) : mg;
    const float bias = isr ? br[ml] : bu[ml];
#pragma unroll
    for (int rt = 0; rt < 2; ++rt)
#pragma unroll
      for (int j = 0; j < 8; ++j) {
        const int n = row0 + wave_r * 32 + rt * 16 + hi * 8 + j;
        const size_t idx = (size_t)n * HDIM + ml;
        const float s = sigmoidf(acc[rt][ct][j] + bias);
        if (!isr) u[idx] = s;
        else      rhx16[idx] = (_Float16)(s * hxf[idx]);
      }
  }
}

// Step 3: k = tanh([x0h,xth,rhx]@W_k + b); hx = (1-u)*k + u*hx (f32 + f16 copy)
__global__ void __launch_bounds__(256)
k_step3(const _Float16* __restrict__ x0h16, const _Float16* __restrict__ xth16,
        const _Float16* __restrict__ rhx16,
        const _Float16* __restrict__ Wt_k, const float* __restrict__ bk,
        const float* __restrict__ u,
        float* __restrict__ hxf, _Float16* __restrict__ hx16)
{
  __shared__ Smem sm;
  const int row0 = blockIdx.x * BLK_R;
  const int col0 = blockIdx.y * BLK_C;
  v8f acc[2][4] = {};
  gemm_tile<3>(x0h16, xth16, rhx16, nullptr, Wt_k, 3 * HDIM, col0, row0, sm, acc);

  const int lane = threadIdx.x & 31, wid = threadIdx.x >> 5;
  const int wave_r = wid & 3, wave_c = wid >> 2;
  const int lh = lane & 15, hi = lane >> 4;
#pragma unroll
  for (int ct = 0; ct < 4; ++ct) {
    const int m = col0 + wave_c * 64 + ct * 16 + lh;
    const float bias = bk[m];
#pragma unroll
    for (int rt = 0; rt < 2; ++rt)
#pragma unroll
      for (int j = 0; j < 8; ++j) {
        const int n = row0 + wave_r * 32 + rt * 16 + hi * 8 + j;
        const size_t idx = (size_t)n * HDIM + m;
        const float kk = tanhf(acc[rt][ct][j] + bias);
        const float uu = u[idx];
        const float h = (1.0f - uu) * kk + uu * hxf[idx];
        hxf[idx]  = h;
        hx16[idx] = (_Float16)h;
      }
  }
}

// ---------------- prep kernels ----------------

// out[m][Ktot] = (f16) W(k, m) with optional two stacked sources (u|r fuse)
__global__ void k_transpose(const float* __restrict__ W0,
                            const float* __restrict__ W1, int msplit, int Ktot,
                            _Float16* __restrict__ out)
{
  const int k = blockIdx.x * 256 + threadIdx.x;
  const int m = blockIdx.y;
  const float* W = (m < msplit) ? W0 : W1;
  const int mm = (m < msplit) ? m : (m - msplit);
  out[(size_t)m * Ktot + k] = (_Float16)W[(size_t)k * HDIM + mm];
}

__global__ void k_init(float* __restrict__ hx, _Float16* __restrict__ hx16,
                       const float* __restrict__ x0f, _Float16* __restrict__ x016,
                       int n4)
{
  const int i = blockIdx.x * 256 + threadIdx.x;
  if (i < n4) {
    ((float4*)hx)[i] = make_float4(0.f, 0.f, 0.f, 0.f);
    v4h z = {};
    ((v4h*)hx16)[i] = z;
    const float4 v = ((const float4*)x0f)[i];
    v4h h; h[0] = (_Float16)v.x; h[1] = (_Float16)v.y;
    h[2] = (_Float16)v.z; h[3] = (_Float16)v.w;
    ((v4h*)x016)[i] = h;
  }
}

__global__ void k_cvt2(const float* __restrict__ a, const float* __restrict__ b,
                       _Float16* __restrict__ ah, _Float16* __restrict__ bh, int n4)
{
  const int i = blockIdx.x * 256 + threadIdx.x;
  if (i < n4) {
    const float4 va = ((const float4*)a)[i];
    v4h h; h[0] = (_Float16)va.x; h[1] = (_Float16)va.y;
    h[2] = (_Float16)va.z; h[3] = (_Float16)va.w;
    ((v4h*)ah)[i] = h;
    const float4 vb = ((const float4*)b)[i];
    v4h g; g[0] = (_Float16)vb.x; g[1] = (_Float16)vb.y;
    g[2] = (_Float16)vb.z; g[3] = (_Float16)vb.w;
    ((v4h*)bh)[i] = g;
  }
}

extern "C" void kernel_launch(void* const* d_in, const int* in_sizes, int n_in,
                              void* d_out, int out_size, void* d_ws, size_t ws_size,
                              hipStream_t stream) {
  const float* edge = (const float*)d_in[0];   // (L, N, H)
  const float* node = (const float*)d_in[1];   // (L+1, N, H)
  const float* Wxi  = (const float*)d_in[2];
  const float* bxi  = (const float*)d_in[3];
  const float* Wu   = (const float*)d_in[4];
  const float* bu   = (const float*)d_in[5];
  const float* Wr   = (const float*)d_in[6];
  const float* br   = (const float*)d_in[7];
  const float* Wk   = (const float*)d_in[8];
  const float* bk   = (const float*)d_in[9];

  float* hxf = (float*)d_out;                  // N*H f32 hx lives in d_out
  const size_t NH = (size_t)NNODES * HDIM;

  char* w = (char*)d_ws;
  _Float16* Wt_xi = (_Float16*)w; w += (size_t)HDIM * 4 * HDIM * 2;
  _Float16* Wt_ur = (_Float16*)w; w += (size_t)2 * HDIM * 3 * HDIM * 2;
  _Float16* Wt_k  = (_Float16*)w; w += (size_t)HDIM * 3 * HDIM * 2;
  _Float16* x016  = (_Float16*)w; w += NH * 2;
  _Float16* hx16  = (_Float16*)w; w += NH * 2;
  _Float16* el16  = (_Float16*)w; w += NH * 2;
  _Float16* xl16  = (_Float16*)w; w += NH * 2;
  _Float16* x0h16 = (_Float16*)w; w += NH * 2;
  _Float16* xth16 = (_Float16*)w; w += NH * 2;
  _Float16* rhx16 = (_Float16*)w; w += NH * 2;
  float*    u     = (float*)w;    w += NH * 4;

  const float* x0f = node + (size_t)NSTEPS * NH;
  const int n4 = (int)(NH / 4);

  const dim3 tb(256);
  // weight prep: Bt[m][k] f16, u|r fused into one 1024-row matrix
  k_transpose<<<dim3(4 * HDIM / 256, HDIM),     tb, 0, stream>>>(Wxi, Wxi, HDIM, 4 * HDIM, Wt_xi);
  k_transpose<<<dim3(3 * HDIM / 256, 2 * HDIM), tb, 0, stream>>>(Wu,  Wr,  HDIM, 3 * HDIM, Wt_ur);
  k_transpose<<<dim3(3 * HDIM / 256, HDIM),     tb, 0, stream>>>(Wk,  Wk,  HDIM, 3 * HDIM, Wt_k);
  k_init<<<(n4 + 255) / 256, tb, 0, stream>>>(hxf, hx16, x0f, x016, n4);

  const dim3 g1(NNODES / BLK_R, HDIM / BLK_C);        // 64 x 4
  const dim3 g2(NNODES / BLK_R, (2 * HDIM) / BLK_C);  // 64 x 8
  const dim3 g3(NNODES / BLK_R, HDIM / BLK_C);        // 64 x 4

  for (int l = 0; l < NSTEPS; ++l) {
    const float* elf = edge + (size_t)l * NH;
    const float* xlf = node + (size_t)l * NH;
    k_cvt2<<<(n4 + 255) / 256, tb, 0, stream>>>(elf, xlf, el16, xl16, n4);
    k_step1<<<g1, tb, 0, stream>>>(hx16, el16, x016, xl16, Wt_xi, bxi,
                                   x0f, xlf, x0h16, xth16);
    k_step2<<<g2, tb, 0, stream>>>(hx16, x0h16, xth16, Wt_ur, bu, br, hxf,
                                   u, rhx16);
    k_step3<<<g3, tb, 0, stream>>>(x0h16, xth16, rhx16, Wt_k, bk, u,
                                   hxf, hx16);
  }
}
